// EnvEmbedding_88356067213765
// MI455X (gfx1250) — matrix operand order repaired and checked by reading, not compiled, not run on hardware
//
#include <hip/hip_runtime.h>

// ---------------------------------------------------------------------------
// VQ-VAE vector quantization for MI455X (gfx1250, wave32, WMMA + TDM).
//   dists[n,k] = ||c_k||^2 - 2 * x_n . c_k   (||x||^2 dropped: row-constant)
//   idx[n]     = argmin_k dists[n,k]
//   out        = (codebook[idx], codebook[idx], idx)
// Matmul in f16 WMMA (f32 accumulate); codebook tile DMA'd to LDS by the
// Tensor Data Mover (tensor_load_to_lds, TENSORcnt).
// ---------------------------------------------------------------------------

typedef __attribute__((ext_vector_type(16))) _Float16 v16h;
typedef __attribute__((ext_vector_type(8)))  _Float16 v8h;
typedef __attribute__((ext_vector_type(8)))  float    v8f;
typedef __attribute__((ext_vector_type(4)))  unsigned int v4u;
typedef __attribute__((ext_vector_type(8)))  int      v8i;
typedef __attribute__((ext_vector_type(4)))  int      v4i;

#define KC    512   // codes
#define DIM   64    // embedding dim
#define ROWS  128   // rows per block (8 waves x 16 rows)
#define NT    256   // threads per block (8 wave32)

// ---- prep: codebook f32 -> f16 (workspace), and ||c_k||^2 ------------------
__global__ __launch_bounds__(NT) void vq_prep(const float* __restrict__ cb,
                                              _Float16* __restrict__ cb16,
                                              float* __restrict__ csqr) {
  const int tid = blockIdx.x * blockDim.x + threadIdx.x;
  if (tid < KC * DIM) cb16[tid] = (_Float16)cb[tid];
  if (tid < KC) {
    const float* row = cb + tid * DIM;
    float s = 0.f;
#pragma unroll
    for (int i = 0; i < DIM; ++i) s += row[i] * row[i];
    csqr[tid] = s;
  }
}

// ---- main: TDM codebook load + WMMA distance matmul + argmin + gather ------
__global__ __launch_bounds__(NT) void vq_main(const float* __restrict__ x,
                                              const float* __restrict__ cb,
                                              const _Float16* __restrict__ cb16,
                                              const float* __restrict__ csqr,
                                              float* __restrict__ out_q,
                                              float* __restrict__ out_qbar,
                                              float* __restrict__ out_idx) {
  __shared__ __align__(32) _Float16 sCode[KC * DIM];   // 64 KB, code-major
  __shared__ __align__(32) _Float16 sX[ROWS * DIM];    // 16 KB, row-major
  __shared__ float sCsqr[KC];                          // 2 KB
  __shared__ int   sRowIdx[ROWS];

  const int  tid     = threadIdx.x;
  const int  lane    = tid & 31;
  const int  wave    = tid >> 5;
  const long rowBase = (long)blockIdx.x * ROWS;

  // Prefetch next block's x tile into cache (speculative -> safe at tail).
  __builtin_prefetch(x + (rowBase + ROWS) * DIM, 0, 1);

  // ---- TDM: DMA the 64 KB f16 codebook global -> LDS (wave 0 issues) ------
  // D# flat copy: data_size = 8 B, tensor_dim0 = tile_dim0 = stride = 8192.
  if (tid < 32) {
    const unsigned long long ga = (unsigned long long)(uintptr_t)cb16;
    const unsigned int ldsA     = (unsigned int)(uintptr_t)&sCode[0];
    v4u g0 = { 1u,                                   // count=1, user mode
               ldsA,                                 // lds_addr
               (unsigned int)(ga & 0xffffffffu),     // global_addr[31:0]
               (unsigned int)((ga >> 32) & 0x01ffffffu) | (2u << 30) }; // +type=2
    v8i g1 = { (int)(3u << 16),        // mask=0, data_size=3 (8B), no pad/iter
               (int)(8192u << 16),     // tensor_dim0[15:0]=8192
               0x00010000,             // tensor_dim0[31:16]=0, tensor_dim1=1
               (int)(8192u << 16),     // tile_dim0=8192
               1,                      // tile_dim1=1, tile_dim2=0
               8192,                   // tensor_dim0_stride[31:0]
               0, 0 };
    v4i gz4 = { 0, 0, 0, 0 };
    v8i gz8 = { 0, 0, 0, 0, 0, 0, 0, 0 };
    __builtin_amdgcn_tensor_load_to_lds(g0, g1, gz4, gz4, gz8, 0);
  }

  // ---- ||c||^2 -> LDS, x tile f32 -> f16 -> LDS (overlaps with TDM) -------
  {
#pragma unroll
    for (int c = 0; c < KC / NT; ++c)                    // 2 iters
      sCsqr[c * NT + tid] = csqr[c * NT + tid];
    const float4* src = (const float4*)(x + rowBase * DIM);
#pragma unroll
    for (int c = 0; c < (ROWS * DIM / 4) / NT; ++c) {    // 8 iters
      const int i = c * NT + tid;
      float4 v = src[i];
      union { _Float16 h[4]; unsigned long long u; } p;
      p.h[0] = (_Float16)v.x; p.h[1] = (_Float16)v.y;
      p.h[2] = (_Float16)v.z; p.h[3] = (_Float16)v.w;
      *(unsigned long long*)&sX[i * 4] = p.u;            // one 8 B ds_store
    }
  }
  if (tid < 32) __builtin_amdgcn_s_wait_tensorcnt(0);    // TDM done (wave 0)
  __syncthreads();

  // A fragments (16x32 f16): lane group 0 (lanes 0-15) holds row
  // (lane&15) with K = k0..k0+15 contiguous, k0=0; group 1: k0=16.
  const int aRow = wave * 16 + (lane & 15);
  const int k0   = (lane < 16) ? 0 : 16;
  v8h a0 = *(const v8h*)&sX[aRow * DIM + k0];
  v8h a1 = *(const v8h*)&sX[aRow * DIM + k0 + 8];
  v8h a2 = *(const v8h*)&sX[aRow * DIM + k0 + 32];
  v8h a3 = *(const v8h*)&sX[aRow * DIM + k0 + 40];
  v16h aLo = __builtin_shufflevector(a0, a1, 0,1,2,3,4,5,6,7,8,9,10,11,12,13,14,15);
  v16h aHi = __builtin_shufflevector(a2, a3, 0,1,2,3,4,5,6,7,8,9,10,11,12,13,14,15);

  // B fragment (32x16 f16): lane n holds column (code) n&15, K = dB..dB+15
  // contiguous -> contiguous 32B per lane from code-major LDS.
  const int dB = (lane < 16) ? 0 : 16;

  float bestD[8];
  int   bestI[8];
#pragma unroll
  for (int j = 0; j < 8; ++j) { bestD[j] = 3.0e38f; bestI[j] = 0; }

#pragma unroll 4
  for (int t = 0; t < KC / 16; ++t) {
    const int code = t * 16 + (lane & 15);
    v16h bLo = *(const v16h*)&sCode[code * DIM + dB];
    v16h bHi = *(const v16h*)&sCode[code * DIM + dB + 32];
    v8f acc = {};
    acc = __builtin_amdgcn_wmma_f32_16x16x32_f16(false, aLo, false, bLo,
                                                 (short)0, acc, false, false);
    acc = __builtin_amdgcn_wmma_f32_16x16x32_f16(false, aHi, false, bHi,
                                                 (short)0, acc, false, false);
    // C/D layout: VGPR j = row j (lanes 0-15) / row j+8 (lanes 16-31),
    // column = lane&15.
    const int   col = t * 16 + (lane & 15);
    const float cs  = sCsqr[col];
#pragma unroll
    for (int j = 0; j < 8; ++j) {
      const float d = cs - 2.0f * acc[j];
      if (d < bestD[j]) { bestD[j] = d; bestI[j] = col; }
    }
  }

  // Min-reduce across the 16 lanes holding the same row (lower-index ties).
#pragma unroll
  for (int j = 0; j < 8; ++j) {
#pragma unroll
    for (int off = 8; off >= 1; off >>= 1) {
      const float od = __shfl_xor(bestD[j], off, 16);
      const int   oi = __shfl_xor(bestI[j], off, 16);
      if (od < bestD[j] || (od == bestD[j] && oi < bestI[j])) {
        bestD[j] = od; bestI[j] = oi;
      }
    }
  }
  if ((lane & 15) == 0) {
    const int g = lane >> 4;
#pragma unroll
    for (int j = 0; j < 8; ++j)
      sRowIdx[wave * 16 + g * 8 + j] = bestI[j];
  }
  __syncthreads();

  // Gather fp32 codebook rows (L2-hot) and stream to both outputs, 128b stores.
  {
    const int  row  = tid >> 1;      // 2 threads per row
    const int  half = tid & 1;
    const long gRow = rowBase + row;
    const int  idx  = sRowIdx[row];
    const float4* q   = (const float4*)(cb + (long)idx * DIM) + half * 8;
    float4*       oq  = (float4*)(out_q    + gRow * DIM) + half * 8;
    float4*       oqb = (float4*)(out_qbar + gRow * DIM) + half * 8;
#pragma unroll
    for (int c = 0; c < 8; ++c) {
      const float4 v = q[c];
      oq[c]  = v;
      oqb[c] = v;
    }
    if (tid < ROWS) out_idx[rowBase + tid] = (float)sRowIdx[tid];
  }
}

extern "C" void kernel_launch(void* const* d_in, const int* in_sizes, int n_in,
                              void* d_out, int out_size, void* d_ws, size_t ws_size,
                              hipStream_t stream) {
  const float* x  = (const float*)d_in[0];   // [N, 64] f32
  const float* cb = (const float*)d_in[1];   // [512, 64] f32
  const int nTotal = in_sizes[0] / DIM;      // 524288

  _Float16* cb16 = (_Float16*)d_ws;                                     // 64 KB
  float*    csqr = (float*)((char*)d_ws + KC * DIM * sizeof(_Float16)); // 2 KB

  float* out_q    = (float*)d_out;
  float* out_qbar = out_q    + (size_t)nTotal * DIM;
  float* out_idx  = out_qbar + (size_t)nTotal * DIM;

  vq_prep<<<(KC * DIM + NT - 1) / NT, NT, 0, stream>>>(cb, cb16, csqr);
  vq_main<<<nTotal / ROWS, NT, 0, stream>>>(x, cb, cb16, csqr,
                                            out_q, out_qbar, out_idx);
}
